// MultiHeadAttention_45689862095325
// MI455X (gfx1250) — matrix operand order repaired and checked by reading, not compiled, not run on hardware
//
#include <hip/hip_runtime.h>
#include <hip/hip_bf16.h>
#include <stdint.h>

// MHA forward for MI455X (gfx1250, wave32).
// bf16 WMMA (v_wmma_f32_16x16x32_bf16, f32 accum) for all four GEMMs +
// flash-attention inner loop. All f32->bf16 conversion hoisted into one
// elementwise pass so GEMM inner loops are pure global_load_b128 + v_wmma.
// Workspace: 64 MB of bf16 staging.

#define TSEQ  4096
#define DM    1024
#define NHEAD 16
#define HD    64

typedef __attribute__((ext_vector_type(16))) __bf16 v16bf;
typedef __attribute__((ext_vector_type(8)))  float  v8f;

union Frag16 {
  v16bf          bf;
  unsigned short us[16];
  uint4          q[2];
};

__device__ __forceinline__ unsigned short f2bf(float f) {
  unsigned int u = __float_as_uint(f);
  u += 0x7FFFu + ((u >> 16) & 1u);          // round-to-nearest-even
  return (unsigned short)(u >> 16);
}

__device__ __forceinline__ v8f wmma_bf16(const Frag16& a, const Frag16& b, v8f c) {
  // D = A(16x32 bf16) x B(32x16 bf16) + C(16x16 f32)
  return __builtin_amdgcn_wmma_f32_16x16x32_bf16(false, a.bf, false, b.bf,
                                                 (short)0, c, false, false);
}

// ---------------------------------------------------------------------------
// Kernel 0: one-shot f32 -> bf16 conversion (vectorized float4 -> uint2).
// ---------------------------------------------------------------------------
__global__ __launch_bounds__(256)
void cvt_kernel(const float* __restrict__ src, unsigned short* __restrict__ dst,
                int n4)
{
  const int i = blockIdx.x * 256 + threadIdx.x;   // one float4 per thread
  if (i < n4) {
    const float4 f = ((const float4*)src)[i];
    uint2 p;
    p.x = (unsigned int)f2bf(f.x) | ((unsigned int)f2bf(f.y) << 16);
    p.y = (unsigned int)f2bf(f.z) | ((unsigned int)f2bf(f.w) << 16);
    ((uint2*)dst)[i] = p;
  }
}

// ---------------------------------------------------------------------------
// Kernel 1: P = X @ W^T + b (torch Linear), bf16 in, bf16 out.
//   mode 0: out[h][t][HD]      (Q, K head-major)
//   mode 1: out[h][HD][TSEQ]   (V transposed: key axis contiguous for PV WMMA)
// Wave tile 32(M) x 64(N): 8 WMMA per K-step from 12 b128 loads.
// 8 waves along M -> WG tile 256 x 64. Grid: (TSEQ/256, DM/64).
// ---------------------------------------------------------------------------
__global__ __launch_bounds__(256)
void proj_kernel(const unsigned short* __restrict__ X,
                 const unsigned short* __restrict__ W,
                 const float* __restrict__ bias,
                 unsigned short* __restrict__ out, int mode)
{
  const int lane = threadIdx.x & 31;
  const int wave = threadIdx.x >> 5;
  const int half = lane >> 4;   // 0: lanes 0-15, 1: lanes 16-31
  const int l16  = lane & 15;

  const int m0 = (blockIdx.x * 8 + wave) * 32;  // row (token) tile base
  const int n0 = blockIdx.y * 64;               // col (feature) tile base

  v8f acc[2][4] = {};

  const unsigned short* arow0 = X + (size_t)(m0 + l16) * DM + half * 8;
  const unsigned short* arow1 = arow0 + (size_t)16 * DM;
  const unsigned short* brow  = W + (size_t)(n0 + l16) * DM + half * 8;

  for (int k0 = 0; k0 < DM; k0 += 32) {
    if (k0 + 32 < DM) {
      __builtin_prefetch(arow0 + k0 + 32, 0, 0);   // global_prefetch_b8
      __builtin_prefetch(arow1 + k0 + 32, 0, 0);
    }
    Frag16 a0, a1;
    a0.q[0] = *(const uint4*)(arow0 + k0);
    a0.q[1] = *(const uint4*)(arow0 + k0 + 16);
    a1.q[0] = *(const uint4*)(arow1 + k0);
    a1.q[1] = *(const uint4*)(arow1 + k0 + 16);
    #pragma unroll
    for (int nt = 0; nt < 4; ++nt) {
      Frag16 b;
      const unsigned short* bp = brow + (size_t)nt * 16 * DM + k0;
      b.q[0] = *(const uint4*)bp;
      b.q[1] = *(const uint4*)(bp + 16);
      acc[0][nt] = wmma_bf16(a0, b, acc[0][nt]);
      acc[1][nt] = wmma_bf16(a1, b, acc[1][nt]);
    }
  }

  // Epilogue: bias + bf16 store.
  #pragma unroll
  for (int g = 0; g < 2; ++g) {
    const int mg = m0 + g * 16;
    #pragma unroll
    for (int nt = 0; nt < 4; ++nt) {
      const int col = n0 + nt * 16 + l16;   // output feature
      const int h   = col >> 6;
      const int d   = col & 63;
      const float bcol = bias[col];
      if (mode == 0) {
        // out[h][t][HD] : per lane 8 strided b16 stores
        #pragma unroll
        for (int r = 0; r < 8; ++r) {
          const int t = mg + half * 8 + r;
          out[((size_t)h * TSEQ + t) * HD + d] = f2bf(acc[g][nt][r] + bcol);
        }
      } else {
        // out[h][d][t] : 8 consecutive t per lane -> one b128 store
        unsigned int p[4];
        #pragma unroll
        for (int i = 0; i < 4; ++i) {
          p[i] = (unsigned int)f2bf(acc[g][nt][2 * i]     + bcol)
               | ((unsigned int)f2bf(acc[g][nt][2 * i + 1] + bcol) << 16);
        }
        const size_t t0 = (size_t)mg + half * 8;
        *(uint4*)(out + ((size_t)h * HD + d) * TSEQ + t0) =
            make_uint4(p[0], p[1], p[2], p[3]);
      }
    }
  }
}

// ---------------------------------------------------------------------------
// Kernel 2: flash attention, causal, per (head, 16-query tile) per wave.
// 32-key blocks: 4 WMMA for S, online softmax (shfl_xor row reductions),
// P C-layout -> A-layout via per-wave LDS tile, 4 WMMA for P @ V.
// Grid: (TSEQ/(16*8), NHEAD), block 256.
// ---------------------------------------------------------------------------
__global__ __launch_bounds__(256)
void attn_kernel(const unsigned short* __restrict__ Qh,
                 const unsigned short* __restrict__ Kh,
                 const unsigned short* __restrict__ VhT,
                 unsigned short* __restrict__ Ctx)
{
  __shared__ unsigned short Pbuf[8][16][32];   // 8 KB, one 16x32 tile per wave

  const int lane = threadIdx.x & 31;
  const int wave = threadIdx.x >> 5;
  const int half = lane >> 4;
  const int l16  = lane & 15;
  const int h    = blockIdx.y;
  const int q0   = (blockIdx.x * 8 + wave) * 16;

  // Q fragments (A layout), two K=32 chunks over head dim 64. Loaded once.
  Frag16 qa[2];
  #pragma unroll
  for (int c = 0; c < 2; ++c) {
    const unsigned short* qp =
        Qh + ((size_t)h * TSEQ + q0 + l16) * HD + c * 32 + half * 8;
    qa[c].q[0] = *(const uint4*)(qp);
    qa[c].q[1] = *(const uint4*)(qp + 16);
  }

  float mrow[8], lrow[8];
  v8f   o[4] = {};
  #pragma unroll
  for (int r = 0; r < 8; ++r) { mrow[r] = -1e30f; lrow[r] = 0.0f; }

  const float scale = 0.125f;   // 1/sqrt(64)
  unsigned short (*P)[32] = Pbuf[wave];

  for (int j0 = 0; j0 < q0 + 16; j0 += 32) {       // causal key-block loop
    // ---- S = Q @ K^T -------------------------------------------------------
    v8f s0 = {}, s1 = {};
    #pragma unroll
    for (int c = 0; c < 2; ++c) {
      Frag16 fk0, fk1;   // B layout: column = key, d in table order (half*8, +16)
      const unsigned short* kp0 =
          Kh + ((size_t)h * TSEQ + j0 + l16) * HD + c * 32 + half * 8;
      const unsigned short* kp1 =
          Kh + ((size_t)h * TSEQ + j0 + 16 + l16) * HD + c * 32 + half * 8;
      fk0.q[0] = *(const uint4*)kp0;  fk0.q[1] = *(const uint4*)(kp0 + 16);
      fk1.q[0] = *(const uint4*)kp1;  fk1.q[1] = *(const uint4*)(kp1 + 16);
      s0 = wmma_bf16(qa[c], fk0, s0);
      s1 = wmma_bf16(qa[c], fk1, s1);
    }

    // ---- scale + causal mask + online softmax ------------------------------
    float mnew[8];
    #pragma unroll
    for (int r = 0; r < 8; ++r) {
      const int qrow = q0 + half * 8 + r;
      float v0 = s0[r] * scale;  if (j0 + l16      > qrow) v0 = -1e30f;
      float v1 = s1[r] * scale;  if (j0 + 16 + l16 > qrow) v1 = -1e30f;
      s0[r] = v0;  s1[r] = v1;
      float mx = fmaxf(v0, v1);
      mx = fmaxf(mx, __shfl_xor(mx, 1));
      mx = fmaxf(mx, __shfl_xor(mx, 2));
      mx = fmaxf(mx, __shfl_xor(mx, 4));
      mx = fmaxf(mx, __shfl_xor(mx, 8));        // row max within 16-lane group
      mnew[r] = fmaxf(mrow[r], mx);
    }
    #pragma unroll
    for (int r = 0; r < 8; ++r) {
      const float p0 = __expf(s0[r] - mnew[r]);
      const float p1 = __expf(s1[r] - mnew[r]);
      s0[r] = p0;  s1[r] = p1;
      float rs = p0 + p1;
      rs += __shfl_xor(rs, 1);
      rs += __shfl_xor(rs, 2);
      rs += __shfl_xor(rs, 4);
      rs += __shfl_xor(rs, 8);                  // row sum
      const float alpha = __expf(mrow[r] - mnew[r]);
      lrow[r] = lrow[r] * alpha + rs;
      mrow[r] = mnew[r];
      #pragma unroll
      for (int dt = 0; dt < 4; ++dt) o[dt][r] *= alpha;
    }

    // ---- C-layout P -> LDS -> A-layout fragment ----------------------------
    #pragma unroll
    for (int r = 0; r < 8; ++r) {
      P[half * 8 + r][l16]      = f2bf(s0[r]);
      P[half * 8 + r][l16 + 16] = f2bf(s1[r]);
    }
    asm volatile("s_wait_dscnt 0" ::: "memory");   // same-wave DS RAW

    Frag16 pa;
    {
      const unsigned short* pp = &P[l16][half * 8];
      pa.q[0] = *(const uint4*)pp;          // keys kb..kb+7
      pa.q[1] = *(const uint4*)(pp + 16);   // keys kb+16..kb+23
    }

    // ---- O += P @ V  (V from VhT: key axis contiguous) ---------------------
    #pragma unroll
    for (int dt = 0; dt < 4; ++dt) {
      Frag16 fv;
      const unsigned short* vp =
          VhT + ((size_t)h * HD + dt * 16 + l16) * TSEQ + j0 + half * 8;
      fv.q[0] = *(const uint4*)vp;  fv.q[1] = *(const uint4*)(vp + 16);
      o[dt] = wmma_bf16(pa, fv, o[dt]);
    }
  }

  // ---- normalize + store context (bf16, token-major for the final GEMM) ----
  unsigned short* cbase = Ctx + (size_t)q0 * DM + h * HD;
  #pragma unroll
  for (int dt = 0; dt < 4; ++dt) {
    #pragma unroll
    for (int r = 0; r < 8; ++r) {
      const float val = o[dt][r] / lrow[r];
      cbase[(size_t)(half * 8 + r) * DM + dt * 16 + l16] = f2bf(val);
    }
  }
}

// ---------------------------------------------------------------------------
// Kernel 3: out = Ctx @ W_o^T + b_o (f32 output). Same 32x64 wave tiling.
// ---------------------------------------------------------------------------
__global__ __launch_bounds__(256)
void oproj_kernel(const unsigned short* __restrict__ Ctx,
                  const unsigned short* __restrict__ W,
                  const float* __restrict__ bias, float* __restrict__ out)
{
  const int lane = threadIdx.x & 31;
  const int wave = threadIdx.x >> 5;
  const int half = lane >> 4;
  const int l16  = lane & 15;

  const int m0 = (blockIdx.x * 8 + wave) * 32;
  const int n0 = blockIdx.y * 64;

  v8f acc[2][4] = {};

  const unsigned short* arow0 = Ctx + (size_t)(m0 + l16) * DM + half * 8;
  const unsigned short* arow1 = arow0 + (size_t)16 * DM;
  const unsigned short* brow  = W + (size_t)(n0 + l16) * DM + half * 8;

  for (int k0 = 0; k0 < DM; k0 += 32) {
    if (k0 + 32 < DM) {
      __builtin_prefetch(arow0 + k0 + 32, 0, 0);
      __builtin_prefetch(arow1 + k0 + 32, 0, 0);
    }
    Frag16 a0, a1;
    a0.q[0] = *(const uint4*)(arow0 + k0);
    a0.q[1] = *(const uint4*)(arow0 + k0 + 16);
    a1.q[0] = *(const uint4*)(arow1 + k0);
    a1.q[1] = *(const uint4*)(arow1 + k0 + 16);
    #pragma unroll
    for (int nt = 0; nt < 4; ++nt) {
      Frag16 b;
      const unsigned short* bp = brow + (size_t)nt * 16 * DM + k0;
      b.q[0] = *(const uint4*)bp;
      b.q[1] = *(const uint4*)(bp + 16);
      acc[0][nt] = wmma_bf16(a0, b, acc[0][nt]);
      acc[1][nt] = wmma_bf16(a1, b, acc[1][nt]);
    }
  }

  #pragma unroll
  for (int g = 0; g < 2; ++g) {
    const int mg = m0 + g * 16;
    #pragma unroll
    for (int nt = 0; nt < 4; ++nt) {
      const int col  = n0 + nt * 16 + l16;
      const float bc = bias[col];
      #pragma unroll
      for (int r = 0; r < 8; ++r) {
        out[(size_t)(mg + half * 8 + r) * DM + col] = acc[g][nt][r] + bc;
      }
    }
  }
}

// ---------------------------------------------------------------------------
extern "C" void kernel_launch(void* const* d_in, const int* in_sizes, int n_in,
                              void* d_out, int out_size, void* d_ws, size_t ws_size,
                              hipStream_t stream) {
  (void)in_sizes; (void)n_in; (void)out_size; (void)ws_size;

  const float* q   = (const float*)d_in[0];
  const float* k   = (const float*)d_in[1];
  const float* v   = (const float*)d_in[2];
  // d_in[3]: causal mask (bool) — causality is computed analytically.
  const float* w_q = (const float*)d_in[4];
  const float* b_q = (const float*)d_in[5];
  const float* w_k = (const float*)d_in[6];
  const float* b_k = (const float*)d_in[7];
  const float* w_v = (const float*)d_in[8];
  const float* b_v = (const float*)d_in[9];
  const float* w_o = (const float*)d_in[10];
  const float* b_o = (const float*)d_in[11];

  // Workspace layout (bf16 halves): 64 MB total.
  const size_t ACT = (size_t)TSEQ * DM;   // 4 Mi elements
  const size_t WT  = (size_t)DM * DM;     // 1 Mi elements
  unsigned short* Qh  = (unsigned short*)d_ws;
  unsigned short* Kh  = Qh  + ACT;
  unsigned short* VhT = Kh  + ACT;
  unsigned short* Ctx = VhT + ACT;
  unsigned short* Xq  = Ctx + ACT;
  unsigned short* Xk  = Xq  + ACT;
  unsigned short* Xv  = Xk  + ACT;
  unsigned short* Wq  = Xv  + ACT;
  unsigned short* Wk  = Wq  + WT;
  unsigned short* Wv  = Wk  + WT;
  unsigned short* Wo  = Wv  + WT;

  // One-shot f32 -> bf16 staging.
  const int actN4 = (int)(ACT / 4), wtN4 = (int)(WT / 4);
  cvt_kernel<<<actN4 / 256, 256, 0, stream>>>(q,   Xq, actN4);
  cvt_kernel<<<actN4 / 256, 256, 0, stream>>>(k,   Xk, actN4);
  cvt_kernel<<<actN4 / 256, 256, 0, stream>>>(v,   Xv, actN4);
  cvt_kernel<<<wtN4  / 256, 256, 0, stream>>>(w_q, Wq, wtN4);
  cvt_kernel<<<wtN4  / 256, 256, 0, stream>>>(w_k, Wk, wtN4);
  cvt_kernel<<<wtN4  / 256, 256, 0, stream>>>(w_v, Wv, wtN4);
  cvt_kernel<<<wtN4  / 256, 256, 0, stream>>>(w_o, Wo, wtN4);

  const dim3 gblk(256);
  const dim3 ggrid(TSEQ / 256, DM / 64);     // 16 x 16

  proj_kernel<<<ggrid, gblk, 0, stream>>>(Xq, Wq, b_q, Qh, 0);
  proj_kernel<<<ggrid, gblk, 0, stream>>>(Xk, Wk, b_k, Kh, 0);
  proj_kernel<<<ggrid, gblk, 0, stream>>>(Xv, Wv, b_v, VhT, 1);

  attn_kernel<<<dim3(TSEQ / 128, NHEAD), gblk, 0, stream>>>(Qh, Kh, VhT, Ctx);

  oproj_kernel<<<ggrid, gblk, 0, stream>>>(Ctx, Wo, b_o, (float*)d_out);
}
